// ScaledDotProductAttention_40785009443047
// MI455X (gfx1250) — matrix operand order repaired and checked by reading, not compiled, not run on hardware
//
#include <hip/hip_runtime.h>
#include <hip/hip_bf16.h>
#include <math.h>

typedef __attribute__((ext_vector_type(16))) _Float16 v16h;
typedef __attribute__((ext_vector_type(8)))  _Float16 v8h;
typedef __attribute__((ext_vector_type(4)))  _Float16 v4h;
typedef __attribute__((ext_vector_type(8)))  float    v8f;
typedef __attribute__((ext_vector_type(4)))  float    v4f;

#define B_    8
#define S_    2048
#define D_    128
#define TKV   32
#define KPAD  136   // K LDS row stride in halves (272B rows: 16B aligned, bank-spread)
#define VPAD  40    // Vt LDS row stride in halves (80B rows)
#define PPAD  36    // P  LDS row stride in halves

// ---------------------------------------------------------------------------
// One-time conversion pass: K -> f16 (natural layout), V -> f16 tile-transposed
// panels Vt[b][s/32][d][kk] so the attention kernel stages contiguous 8KB tiles.
// ---------------------------------------------------------------------------
__global__ __launch_bounds__(256)
void cvt_k_f16(const float* __restrict__ k, _Float16* __restrict__ kh)
{
    const size_t i = ((size_t)blockIdx.x * 256 + threadIdx.x) * 8;
    v4f a = *(const v4f*)(k + i);
    v4f b = *(const v4f*)(k + i + 4);
    v8h o;
    o[0] = (_Float16)a.x; o[1] = (_Float16)a.y; o[2] = (_Float16)a.z; o[3] = (_Float16)a.w;
    o[4] = (_Float16)b.x; o[5] = (_Float16)b.y; o[6] = (_Float16)b.z; o[7] = (_Float16)b.w;
    *(v8h*)(kh + i) = o;
}

__global__ __launch_bounds__(256)
void cvt_v_f16t(const float* __restrict__ v, _Float16* __restrict__ vt)
{
    // output linear o over layout ((b*(S/32) + sb)*D + d)*32 + kk
    const size_t o  = ((size_t)blockIdx.x * 256 + threadIdx.x) * 8;
    const int    kk = (int)(o & 31);
    const size_t r1 = o >> 5;
    const int    d  = (int)(r1 & (D_ - 1));
    const size_t bs = r1 >> 7;                 // b*(S/32) + sb
    const int    sb = (int)(bs & (S_ / TKV - 1));
    const size_t b  = bs / (S_ / TKV);
    const float* src = v + ((b * S_) + (size_t)sb * TKV + kk) * D_ + d;
    v8h oh;
    #pragma unroll
    for (int e = 0; e < 8; ++e) oh[e] = (_Float16)src[(size_t)e * D_];
    *(v8h*)(vt + o) = oh;
}

// ---------------------------------------------------------------------------
// Main kernel: 8 waves / 256 threads, q-tile 128 (16 q rows per wave),
// double-buffered f16 K/V tiles in LDS, online-softmax flash attention.
// ---------------------------------------------------------------------------
__global__ __launch_bounds__(256)
void sdpa_flash_wmma_f16kv(const float* __restrict__ q,
                           const _Float16* __restrict__ kh,
                           const _Float16* __restrict__ vth,
                           const unsigned char* __restrict__ mask,
                           float* __restrict__ out)
{
    __shared__ _Float16 Klds[2][TKV][KPAD];   // [buf][kv][d]
    __shared__ _Float16 Vt[2][D_][VPAD];      // [buf][d][kv]
    __shared__ _Float16 Plds[8][16][PPAD];    // per-wave P staging

    const int tid  = threadIdx.x;
    const int lane = tid & 31;
    const int wave = tid >> 5;
    const int lrow = lane & 15;
    const int uh   = (lane >> 4) & 1;

    const int ntiles = S_ / 128;
    const int b  = blockIdx.x / ntiles;
    const int qt = blockIdx.x % ntiles;
    const int q0 = qt * 128 + wave * 16;

    const float scale = 0.08838834764831845f;   // 1/sqrt(128)
    const float LOG2E = 1.4426950408889634f;

    // Q fragments (A layout), 4 chunks of K=32 over D=128
    v16h qf[4];
    {
        const float* qrow = q + (size_t)(b * S_ + q0 + lrow) * D_;
        #pragma unroll
        for (int c = 0; c < 4; ++c) {
            const int d0 = c * 32 + uh * 8;
            #pragma unroll
            for (int e = 0; e < 8; ++e) {
                qf[c][e]     = (_Float16)qrow[d0 + e];
                qf[c][8 + e] = (_Float16)qrow[d0 + 16 + e];
            }
        }
    }

    v8f acc[8];
    #pragma unroll
    for (int f = 0; f < 8; ++f) acc[f] = {};
    float mrun[8], lrun[8];
    #pragma unroll
    for (int j = 0; j < 8; ++j) { mrun[j] = -INFINITY; lrun[j] = 0.0f; }

    auto stage = [&](int buf, int kv) {
        // K tile: contiguous 8KB (32 rows x 256B), 32B per thread
        const _Float16* ksrc = kh + (size_t)(b * S_ + kv) * D_;
        {
            const int row = tid >> 3;
            const int c   = (tid & 7) * 16;
            const v8h* s0 = (const v8h*)(ksrc + (size_t)row * D_ + c);
            *(v8h*)&Klds[buf][row][c]     = s0[0];
            *(v8h*)&Klds[buf][row][c + 8] = s0[1];
        }
        // V tile: contiguous 8KB transposed panel, 32B per thread
        const _Float16* vsrc = vth + (size_t)(b * (S_ / TKV) + kv / TKV) * (D_ * TKV);
        {
            const int row = tid >> 1;
            const int c   = (tid & 1) * 16;
            const v8h* s1 = (const v8h*)(vsrc + (size_t)row * TKV + c);
            *(v8h*)&Vt[buf][row][c]     = s1[0];
            *(v8h*)&Vt[buf][row][c + 8] = s1[1];
        }
    };

    stage(0, 0);

    for (int kv0 = 0; kv0 < S_; kv0 += TKV) {
        const int buf = (kv0 >> 5) & 1;
        __syncthreads();                          // staging of `buf` visible to all
        if (kv0 + TKV < S_) stage(buf ^ 1, kv0 + TKV);

        // ---- S = Q * K^T ----
        v8f s[2];
        #pragma unroll
        for (int t = 0; t < 2; ++t) {
            v8f c = {};
            #pragma unroll
            for (int cc = 0; cc < 4; ++cc) {
                const _Float16* src = &Klds[buf][t * 16 + lrow][cc * 32 + uh * 8];
                v8h lo = *(const v8h*)(src);
                v8h hi = *(const v8h*)(src + 16);
                v16h bk;
                #pragma unroll
                for (int e = 0; e < 8; ++e) { bk[e] = lo[e]; bk[8 + e] = hi[e]; }
                c = __builtin_amdgcn_wmma_f32_16x16x32_f16(false, qf[cc], false, bk,
                                                           (short)0, c, false, false);
            }
            s[t] = c;
        }

        // ---- scale + mask ----
        #pragma unroll
        for (int t = 0; t < 2; ++t) {
            #pragma unroll
            for (int j = 0; j < 8; ++j) {
                const float sv = s[t][j] * scale;
                const unsigned char mb =
                    mask[((size_t)b * S_ + (q0 + uh * 8 + j)) * S_ + kv0 + t * 16 + lrow];
                s[t][j] = mb ? sv : -INFINITY;
            }
        }

        // ---- online softmax ----
        #pragma unroll
        for (int j = 0; j < 8; ++j) {
            float vmax = fmaxf(s[0][j], s[1][j]);
            #pragma unroll
            for (int off = 8; off >= 1; off >>= 1)
                vmax = fmaxf(vmax, __shfl_xor(vmax, off, 16));
            const float mnew  = fmaxf(mrun[j], vmax);
            const float alpha = exp2f((mrun[j] - mnew) * LOG2E);
            mrun[j] = mnew;
            const float p0 = exp2f((s[0][j] - mnew) * LOG2E);
            const float p1 = exp2f((s[1][j] - mnew) * LOG2E);
            s[0][j] = p0; s[1][j] = p1;
            float rs = p0 + p1;
            #pragma unroll
            for (int off = 8; off >= 1; off >>= 1)
                rs += __shfl_xor(rs, off, 16);
            lrun[j] = lrun[j] * alpha + rs;
            #pragma unroll
            for (int f = 0; f < 8; ++f) acc[f][j] *= alpha;
        }

        // ---- P: C-layout -> A-layout via per-wave LDS ----
        #pragma unroll
        for (int t = 0; t < 2; ++t)
            #pragma unroll
            for (int j = 0; j < 8; ++j)
                Plds[wave][uh * 8 + j][t * 16 + lrow] = (_Float16)s[t][j];

        asm volatile("s_wait_dscnt 0" ::: "memory");

        v16h pf;
        {
            const _Float16* pr = &Plds[wave][lrow][0];
            #pragma unroll
            for (int e = 0; e < 8; ++e) {
                pf[e]     = pr[uh * 8 + e];
                pf[8 + e] = pr[16 + uh * 8 + e];
            }
        }

        // ---- O += P * V ----
        #pragma unroll
        for (int f = 0; f < 8; ++f) {
            const _Float16* vb = &Vt[buf][f * 16 + lrow][uh * 8];
            v8h lo = *(const v8h*)(vb);
            v8h hi = *(const v8h*)(vb + 16);
            v16h bv;
            #pragma unroll
            for (int e = 0; e < 8; ++e) { bv[e] = lo[e]; bv[8 + e] = hi[e]; }
            acc[f] = __builtin_amdgcn_wmma_f32_16x16x32_f16(false, pf, false, bv,
                                                            (short)0, acc[f], false, false);
        }
    }

    #pragma unroll
    for (int j = 0; j < 8; ++j) {
        const float inv = 1.0f / lrun[j];
        const size_t orow = (size_t)(b * S_ + q0 + uh * 8 + j) * D_;
        #pragma unroll
        for (int f = 0; f < 8; ++f)
            out[orow + f * 16 + lrow] = acc[f][j] * inv;
    }
}

// ---------------------------------------------------------------------------
// Fallback (round-1 kernel): fp32 K/V staged + converted in-kernel, 4 waves.
// Used only if the workspace cannot hold the f16 K/V copies.
// ---------------------------------------------------------------------------
__global__ __launch_bounds__(128)
void sdpa_flash_wmma_fp32kv(const float* __restrict__ q, const float* __restrict__ k,
                            const float* __restrict__ v, const unsigned char* __restrict__ mask,
                            float* __restrict__ out)
{
    __shared__ _Float16 Klds[TKV][KPAD];
    __shared__ _Float16 Vt[D_][VPAD];
    __shared__ _Float16 Plds[4][16][PPAD];

    const int tid  = threadIdx.x;
    const int lane = tid & 31;
    const int wave = tid >> 5;
    const int lrow = lane & 15;
    const int uh   = (lane >> 4) & 1;

    const int ntiles = S_ / 64;
    const int b  = blockIdx.x / ntiles;
    const int qt = blockIdx.x % ntiles;
    const int q0 = qt * 64 + wave * 16;

    const float scale = 0.08838834764831845f;
    const float LOG2E = 1.4426950408889634f;

    v16h qf[4];
    {
        const float* qrow = q + (size_t)(b * S_ + q0 + lrow) * D_;
        #pragma unroll
        for (int c = 0; c < 4; ++c) {
            const int d0 = c * 32 + uh * 8;
            #pragma unroll
            for (int e = 0; e < 8; ++e) {
                qf[c][e]     = (_Float16)qrow[d0 + e];
                qf[c][8 + e] = (_Float16)qrow[d0 + 16 + e];
            }
        }
    }

    v8f acc[8];
    #pragma unroll
    for (int f = 0; f < 8; ++f) acc[f] = {};
    float mrun[8], lrun[8];
    #pragma unroll
    for (int j = 0; j < 8; ++j) { mrun[j] = -INFINITY; lrun[j] = 0.0f; }

    for (int kv0 = 0; kv0 < S_; kv0 += TKV) {
        __syncthreads();
        #pragma unroll
        for (int it = 0; it < 8; ++it) {
            const int idx4 = it * 128 + tid;
            const int row  = idx4 >> 5;
            const int col  = (idx4 & 31) * 4;
            const size_t g = (size_t)(b * S_ + kv0 + row) * D_ + col;
            v4f kd = *(const v4f*)(k + g);
            v4f vd = *(const v4f*)(v + g);
            v4h khh; khh.x = (_Float16)kd.x; khh.y = (_Float16)kd.y;
                     khh.z = (_Float16)kd.z; khh.w = (_Float16)kd.w;
            *(v4h*)&Klds[row][col] = khh;
            Vt[col + 0][row] = (_Float16)vd.x;
            Vt[col + 1][row] = (_Float16)vd.y;
            Vt[col + 2][row] = (_Float16)vd.z;
            Vt[col + 3][row] = (_Float16)vd.w;
        }
        __syncthreads();

        v8f s[2];
        #pragma unroll
        for (int t = 0; t < 2; ++t) {
            v8f c = {};
            #pragma unroll
            for (int cc = 0; cc < 4; ++cc) {
                const _Float16* src = &Klds[t * 16 + lrow][cc * 32 + uh * 8];
                v8h lo = *(const v8h*)(src);
                v8h hi = *(const v8h*)(src + 16);
                v16h bk;
                #pragma unroll
                for (int e = 0; e < 8; ++e) { bk[e] = lo[e]; bk[8 + e] = hi[e]; }
                c = __builtin_amdgcn_wmma_f32_16x16x32_f16(false, qf[cc], false, bk,
                                                           (short)0, c, false, false);
            }
            s[t] = c;
        }

        #pragma unroll
        for (int t = 0; t < 2; ++t)
            #pragma unroll
            for (int j = 0; j < 8; ++j) {
                const float sv = s[t][j] * scale;
                const unsigned char mb =
                    mask[((size_t)b * S_ + (q0 + uh * 8 + j)) * S_ + kv0 + t * 16 + lrow];
                s[t][j] = mb ? sv : -INFINITY;
            }

        #pragma unroll
        for (int j = 0; j < 8; ++j) {
            float vmax = fmaxf(s[0][j], s[1][j]);
            #pragma unroll
            for (int off = 8; off >= 1; off >>= 1)
                vmax = fmaxf(vmax, __shfl_xor(vmax, off, 16));
            const float mnew  = fmaxf(mrun[j], vmax);
            const float alpha = exp2f((mrun[j] - mnew) * LOG2E);
            mrun[j] = mnew;
            const float p0 = exp2f((s[0][j] - mnew) * LOG2E);
            const float p1 = exp2f((s[1][j] - mnew) * LOG2E);
            s[0][j] = p0; s[1][j] = p1;
            float rs = p0 + p1;
            #pragma unroll
            for (int off = 8; off >= 1; off >>= 1)
                rs += __shfl_xor(rs, off, 16);
            lrun[j] = lrun[j] * alpha + rs;
            #pragma unroll
            for (int f = 0; f < 8; ++f) acc[f][j] *= alpha;
        }

        #pragma unroll
        for (int t = 0; t < 2; ++t)
            #pragma unroll
            for (int j = 0; j < 8; ++j)
                Plds[wave][uh * 8 + j][t * 16 + lrow] = (_Float16)s[t][j];

        asm volatile("s_wait_dscnt 0" ::: "memory");

        v16h pf;
        {
            const _Float16* pr = &Plds[wave][lrow][0];
            #pragma unroll
            for (int e = 0; e < 8; ++e) {
                pf[e]     = pr[uh * 8 + e];
                pf[8 + e] = pr[16 + uh * 8 + e];
            }
        }

        #pragma unroll
        for (int f = 0; f < 8; ++f) {
            const _Float16* vb = &Vt[f * 16 + lrow][uh * 8];
            v8h lo = *(const v8h*)(vb);
            v8h hi = *(const v8h*)(vb + 16);
            v16h bv;
            #pragma unroll
            for (int e = 0; e < 8; ++e) { bv[e] = lo[e]; bv[8 + e] = hi[e]; }
            acc[f] = __builtin_amdgcn_wmma_f32_16x16x32_f16(false, pf, false, bv,
                                                            (short)0, acc[f], false, false);
        }
    }

    #pragma unroll
    for (int j = 0; j < 8; ++j) {
        const float inv = 1.0f / lrun[j];
        const size_t orow = (size_t)(b * S_ + q0 + uh * 8 + j) * D_;
        #pragma unroll
        for (int f = 0; f < 8; ++f)
            out[orow + f * 16 + lrow] = acc[f][j] * inv;
    }
}

extern "C" void kernel_launch(void* const* d_in, const int* in_sizes, int n_in,
                              void* d_out, int out_size, void* d_ws, size_t ws_size,
                              hipStream_t stream) {
    const float*         q    = (const float*)d_in[0];
    const float*         k    = (const float*)d_in[1];
    const float*         v    = (const float*)d_in[2];
    const unsigned char* mask = (const unsigned char*)d_in[3];
    float*               out  = (float*)d_out;

    const size_t elems = (size_t)B_ * S_ * D_;          // 2M per tensor
    const size_t need  = elems * sizeof(_Float16) * 2;  // f16 K + f16 V

    if (ws_size >= need) {
        _Float16* kh = (_Float16*)d_ws;
        _Float16* vt = kh + elems;
        const int cvt_blocks = (int)(elems / (256 * 8)); // 1024
        cvt_k_f16 <<<cvt_blocks, 256, 0, stream>>>(k, kh);
        cvt_v_f16t<<<cvt_blocks, 256, 0, stream>>>(v, vt);
        dim3 grid(B_ * (S_ / 128));                      // 128 workgroups
        sdpa_flash_wmma_f16kv<<<grid, 256, 0, stream>>>(q, kh, vt, mask, out);
    } else {
        dim3 grid(B_ * (S_ / 64));                       // 256 workgroups
        sdpa_flash_wmma_fp32kv<<<grid, 128, 0, stream>>>(q, k, v, mask, out);
    }
}
